// MinkResBlock_v2_51144470561089
// MI455X (gfx1250) — matrix operand order repaired and checked by reading, not compiled, not run on hardware
//
#include <hip/hip_runtime.h>

// ---------------- problem constants ----------------
#define NVOX   400000
#define CIN    256
#define COUT   256
#define D2     64
#define KOFF   125
#define EPS    1e-5f

// ---------------- vector types ----------------
typedef __attribute__((ext_vector_type(2)))  float    v2f;
typedef __attribute__((ext_vector_type(8)))  float    v8f;
typedef __attribute__((ext_vector_type(16))) __bf16   v16bf;
typedef __attribute__((ext_vector_type(4)))  unsigned v4u;
typedef __attribute__((ext_vector_type(8)))  int      v8i;
typedef __attribute__((ext_vector_type(4)))  int      v4i;

// ---------------- workspace layout (bytes) ----------------
// stats (float index):
//   [0,64)sum1 [64,128)ssq1 [128,384)sum2 [384,640)ssq2
//   [640,704)scale1 [704,768)bias1 [768,1024)scale2 [1024,1280)bias2
#define OFF_STATS 0
#define OFF_WKT   (8192)
#define OFF_H1    (OFF_WKT + 1048576)                 // WkT bf16: 125*64*64*2 (padded)
#define OFF_YA    (OFF_H1 + 51200000 + 56768)         // h1 bf16: N*64*2 (padded)
// yA: N*64*4 bytes (y1 f32, later reused as h2 f32)

__device__ __forceinline__ unsigned short f32_bf16(float f) {
    unsigned u = __builtin_bit_cast(unsigned, f);
    u += 0x7FFFu + ((u >> 16) & 1u);                  // round-to-nearest-even
    return (unsigned short)(u >> 16);
}

__device__ __forceinline__ v16bf load_a16(const unsigned short* p) {
    // A operand in this code's contiguous k order (B matches): lane holds one contiguous 16-elem run (K..K+15)
    union { uint4 u[2]; v16bf v; } t;
    t.u[0] = *(const uint4*)(p);
    t.u[1] = *(const uint4*)(p + 8);
    return t.v;
}

// TDM: DMA one 8KB weight tile (1024 x 8B) from global into LDS at lds_off.
// D# per CDNA5 ISA ch.8: group0 {count=1, lds_addr, global_addr[56:0], type=2},
// group1 {data_size=3(8B), tensor_dim0=1024, tensor_dim1=1, tile_dim0=1024,
//         tile_dim1=1, tensor_dim0_stride=1024}, groups 2/3 zero (<=2D).
// This toolchain declares the 6-arg builtin form (extra int32x8 group).
__device__ __forceinline__ void tdm_load_tile(const unsigned short* src, unsigned lds_off) {
    unsigned long long ga = (unsigned long long)(const void*)src;
    v4u g0;
    g0[0] = 1u;                                   // count=1, user descriptor
    g0[1] = lds_off;                              // lds_addr (bytes)
    g0[2] = (unsigned)ga;                         // global_addr[31:0]
    g0[3] = (unsigned)(ga >> 32) | (2u << 30);    // global_addr[56:32] | type=2
    v8i g1;
    g1[0] = (int)(3u << 16);                      // data_size=3 (8 bytes)
    g1[1] = (int)(1024u << 16);                   // tensor_dim0[15:0]=1024
    g1[2] = (int)(1u << 16);                      // tensor_dim0 hi=0 | tensor_dim1=1
    g1[3] = (int)(1024u << 16);                   // tensor_dim1 hi=0 | tile_dim0=1024
    g1[4] = 1;                                    // tile_dim1=1, tile_dim2=0
    g1[5] = 1024;                                 // tensor_dim0_stride
    g1[6] = 0;
    g1[7] = 0;
    v4i z4 = {0, 0, 0, 0};
    v8i z8 = {0, 0, 0, 0, 0, 0, 0, 0};
    __builtin_amdgcn_tensor_load_to_lds(g0, g1, z4, z4, z8, 0);
}

// ---------------- k0: zero stat accumulators ----------------
__global__ void k_zero(float* stats) {
    int t = blockIdx.x * blockDim.x + threadIdx.x;
    if (t < 640) stats[t] = 0.0f;
}

// ---------------- k1: Wk f32 [125][64][64] -> WkT bf16 [125][n=64][k=64] ----------------
__global__ void k_cvt_wkT(const float* __restrict__ Wk, unsigned short* __restrict__ wkT) {
    int t = blockIdx.x * blockDim.x + threadIdx.x;
    if (t >= KOFF * 64 * 64) return;
    int k = t >> 12;
    int i = (t >> 6) & 63;   // input-channel (K of GEMM)
    int j = t & 63;          // output-channel (N of GEMM)
    wkT[(k << 12) + (j << 6) + i] = f32_bf16(Wk[t]);
}

// ---------------- k2: y1 = x @ W1 (f32 WMMA) + BN1 stats ----------------
__global__ __launch_bounds__(256) void k_gemm1(const float* __restrict__ x,
                                               const float* __restrict__ W1,
                                               float* __restrict__ y,
                                               float* __restrict__ stats) {
    __shared__ float w1s[CIN * D2];          // 64 KB, pair-interleaved [(k/2)][n][k%2]
    int tid = threadIdx.x;
    for (int i = tid; i < CIN * D2; i += 256) {
        int kk = i >> 6, n = i & 63;
        w1s[(kk >> 1) * 128 + n * 2 + (kk & 1)] = W1[i];
    }
    __syncthreads();

    int lane = tid & 31, wave = tid >> 5;
    int m = lane & 15, half = lane >> 4;
    int rb = blockIdx.x * 128 + wave * 16;
    int row = rb + m;

    v8f acc[4] = {};
    const float* xrow = x + (size_t)row * CIN + 2 * half;

    for (int k0 = 0; k0 < CIN; k0 += 4) {
        v2f a; a.x = xrow[k0]; a.y = xrow[k0 + 1];
        int pairbase = ((k0 >> 1) + half) * 128;
#pragma unroll
        for (int ct = 0; ct < 4; ++ct) {
            int n = ct * 16 + m;
            v2f b = *(const v2f*)&w1s[pairbase + n * 2];   // single ds_load_b64
            acc[ct] = __builtin_amdgcn_wmma_f32_16x16x4_f32(false, a, false, b,
                                                            (short)0, acc[ct], false, false);
        }
    }

    float ps[4], pq[4];
#pragma unroll
    for (int ct = 0; ct < 4; ++ct) {
        int col = ct * 16 + m;
        float s = 0.f, q = 0.f;
#pragma unroll
        for (int r = 0; r < 8; ++r) {
            float v = acc[ct][r];
            int orow = rb + r + half * 8;
            y[(size_t)orow * D2 + col] = v;
            s += v; q += v * v;
        }
        ps[ct] = s; pq[ct] = q;
    }

    __syncthreads();
    if (tid < 128) w1s[tid] = 0.0f;          // reuse LDS: bsum[64] | bssq[64]
    __syncthreads();
#pragma unroll
    for (int ct = 0; ct < 4; ++ct) {
        int col = ct * 16 + m;
        atomicAdd(&w1s[col], ps[ct]);
        atomicAdd(&w1s[64 + col], pq[ct]);
    }
    __syncthreads();
    if (tid < 64) {
        atomicAdd(&stats[tid], w1s[tid]);
        atomicAdd(&stats[64 + tid], w1s[64 + tid]);
    }
}

// ---------------- k3: finalize BN1 ----------------
__global__ void k_fin1(const float* __restrict__ g, const float* __restrict__ b,
                       float* __restrict__ stats) {
    int t = threadIdx.x;
    if (t < 64) {
        float inv_n = 1.0f / (float)NVOX;
        float mean = stats[t] * inv_n;
        float var  = stats[64 + t] * inv_n - mean * mean;
        float rs   = rsqrtf(var + EPS);
        float sc   = g[t] * rs;
        stats[640 + t] = sc;
        stats[704 + t] = b[t] - mean * sc;
    }
}

// ---------------- k4: h1 = bf16(relu(y1*scale+bias)) ----------------
__global__ __launch_bounds__(256) void k_bn1(const float* __restrict__ y,
                                             const float* __restrict__ stats,
                                             unsigned short* __restrict__ h1) {
    size_t t = (size_t)blockIdx.x * 256 + threadIdx.x;
    size_t base = t * 4;
    if (base >= (size_t)NVOX * D2) return;
    float4 v = *(const float4*)(y + base);
    int c = (int)(base & 63);
    float o0 = fmaxf(v.x * stats[640 + c]     + stats[704 + c],     0.f);
    float o1 = fmaxf(v.y * stats[640 + c + 1] + stats[704 + c + 1], 0.f);
    float o2 = fmaxf(v.z * stats[640 + c + 2] + stats[704 + c + 2], 0.f);
    float o3 = fmaxf(v.w * stats[640 + c + 3] + stats[704 + c + 3], 0.f);
    uint2 r;
    r.x = (unsigned)f32_bf16(o0) | ((unsigned)f32_bf16(o1) << 16);
    r.y = (unsigned)f32_bf16(o2) | ((unsigned)f32_bf16(o3) << 16);
    *(uint2*)(h1 + base) = r;
}

// ---------------- k5: spconv  h2 = sum_k gather(h1,nbr[k]) @ Wk[k]  (bf16 WMMA + TDM) -------
__global__ __launch_bounds__(256) void k_spconv(const unsigned short* __restrict__ h1,
                                                const unsigned short* __restrict__ wkT,
                                                const int* __restrict__ nbr,
                                                float* __restrict__ h2) {
    __shared__ unsigned short wks[2 * 4096];   // 16 KB: double-buffered bf16 weight tiles
    int tid = threadIdx.x, lane = tid & 31, wave = tid >> 5;
    int m = lane & 15, half = lane >> 4;
    int rb = blockIdx.x * 128 + wave * 16;
    int koffA = half * 16;     // A-matrix lane K offset (channels)
    int khB   = half * 16;     // B-matrix lane K offset

    v8f acc[4] = {};

    // prime buffer 0 via Tensor Data Mover (wave 0 issues, TENSORcnt is per-wave)
    if (wave == 0) {
        tdm_load_tile(wkT, 0);
        __builtin_amdgcn_s_wait_tensorcnt(0);
    }
    __syncthreads();

    for (int k = 0; k < KOFF; ++k) {
        // stream next tile into the other buffer while everyone computes this one
        if (wave == 0 && (k + 1) < KOFF)
            tdm_load_tile(wkT + (size_t)(k + 1) * 4096, (unsigned)(((k + 1) & 1) * 8192));

        const unsigned short* wb = wks + (k & 1) * 4096;

        int idx = nbr[(size_t)k * NVOX + rb + m];
        const unsigned short* arow = h1 + (size_t)idx * D2;
        v16bf a0 = load_a16(arow + koffA);        // channels 0..31
        v16bf a1 = load_a16(arow + 32 + koffA);   // channels 32..63

#pragma unroll
        for (int ct = 0; ct < 4; ++ct) {
            int n = ct * 16 + m;
            v16bf b0 = *(const v16bf*)(wb + n * 64 + khB);
            v16bf b1 = *(const v16bf*)(wb + n * 64 + 32 + khB);
            acc[ct] = __builtin_amdgcn_wmma_f32_16x16x32_bf16(false, a0, false, b0,
                                                              (short)0, acc[ct], false, false);
            acc[ct] = __builtin_amdgcn_wmma_f32_16x16x32_bf16(false, a1, false, b1,
                                                              (short)0, acc[ct], false, false);
        }

        if (wave == 0)
            __builtin_amdgcn_s_wait_tensorcnt(0);   // next tile landed
        __syncthreads();                            // publish to all waves
    }

#pragma unroll
    for (int ct = 0; ct < 4; ++ct) {
        int col = ct * 16 + m;
#pragma unroll
        for (int r = 0; r < 8; ++r) {
            int orow = rb + r + half * 8;
            h2[(size_t)orow * D2 + col] = acc[ct][r];
        }
    }
}

// ---------------- k6: BN2 stats of h2 @ W2 (no store) ----------------
__global__ __launch_bounds__(256) void k_gemm2_stats(const float* __restrict__ h2,
                                                     const float* __restrict__ W2,
                                                     float* __restrict__ stats) {
    __shared__ float w2s[D2 * COUT];          // 64 KB, pair-interleaved [(k/2)][n][k%2]
    int tid = threadIdx.x;
    for (int i = tid; i < D2 * COUT; i += 256) {
        int kk = i >> 8, n = i & 255;
        w2s[(kk >> 1) * 512 + n * 2 + (kk & 1)] = W2[i];
    }
    __syncthreads();

    int lane = tid & 31, wave = tid >> 5;
    int m = lane & 15, half = lane >> 4;
    int rb = blockIdx.x * 128 + wave * 16;
    int row = rb + m;

    v8f acc[16] = {};
    const float* arow = h2 + (size_t)row * D2 + 2 * half;

    for (int k0 = 0; k0 < D2; k0 += 4) {
        v2f a; a.x = arow[k0]; a.y = arow[k0 + 1];
        int pairbase = ((k0 >> 1) + half) * 512;
#pragma unroll
        for (int ct = 0; ct < 16; ++ct) {
            int n = ct * 16 + m;
            v2f b = *(const v2f*)&w2s[pairbase + n * 2];
            acc[ct] = __builtin_amdgcn_wmma_f32_16x16x4_f32(false, a, false, b,
                                                            (short)0, acc[ct], false, false);
        }
    }

    float ps[16], pq[16];
#pragma unroll
    for (int ct = 0; ct < 16; ++ct) {
        float s = 0.f, q = 0.f;
#pragma unroll
        for (int r = 0; r < 8; ++r) { float v = acc[ct][r]; s += v; q += v * v; }
        ps[ct] = s; pq[ct] = q;
    }

    __syncthreads();
    if (tid < 512) w2s[tid] = 0.0f;           // reuse LDS: bsum[256] | bssq[256]
    __syncthreads();
#pragma unroll
    for (int ct = 0; ct < 16; ++ct) {
        int col = ct * 16 + m;
        atomicAdd(&w2s[col], ps[ct]);
        atomicAdd(&w2s[256 + col], pq[ct]);
    }
    __syncthreads();
    if (tid < 256) {
        atomicAdd(&stats[128 + tid], w2s[tid]);
        atomicAdd(&stats[384 + tid], w2s[256 + tid]);
    }
}

// ---------------- k7: finalize BN2 ----------------
__global__ void k_fin2(const float* __restrict__ g, const float* __restrict__ b,
                       float* __restrict__ stats) {
    int t = threadIdx.x;
    if (t < 256) {
        float inv_n = 1.0f / (float)NVOX;
        float mean = stats[128 + t] * inv_n;
        float var  = stats[384 + t] * inv_n - mean * mean;
        float rs   = rsqrtf(var + EPS);
        float sc   = g[t] * rs;
        stats[768 + t]  = sc;
        stats[1024 + t] = b[t] - mean * sc;
    }
}

// ---------------- k8: out = relu(bn2(h2@W2)) + x ----------------
__global__ __launch_bounds__(256) void k_final(const float* __restrict__ h2,
                                               const float* __restrict__ W2,
                                               const float* __restrict__ x,
                                               const float* __restrict__ stats,
                                               float* __restrict__ out) {
    __shared__ float w2s[D2 * COUT];          // 64 KB, pair-interleaved
    int tid = threadIdx.x;
    for (int i = tid; i < D2 * COUT; i += 256) {
        int kk = i >> 8, n = i & 255;
        w2s[(kk >> 1) * 512 + n * 2 + (kk & 1)] = W2[i];
    }
    __syncthreads();

    int lane = tid & 31, wave = tid >> 5;
    int m = lane & 15, half = lane >> 4;
    int rb = blockIdx.x * 128 + wave * 16;
    int row = rb + m;

    v8f acc[16] = {};
    const float* arow = h2 + (size_t)row * D2 + 2 * half;

    for (int k0 = 0; k0 < D2; k0 += 4) {
        v2f a; a.x = arow[k0]; a.y = arow[k0 + 1];
        int pairbase = ((k0 >> 1) + half) * 512;
#pragma unroll
        for (int ct = 0; ct < 16; ++ct) {
            int n = ct * 16 + m;
            v2f b = *(const v2f*)&w2s[pairbase + n * 2];
            acc[ct] = __builtin_amdgcn_wmma_f32_16x16x4_f32(false, a, false, b,
                                                            (short)0, acc[ct], false, false);
        }
    }

#pragma unroll
    for (int ct = 0; ct < 16; ++ct) {
        int c = ct * 16 + m;
        float sc = stats[768 + c], bi = stats[1024 + c];
#pragma unroll
        for (int r = 0; r < 8; ++r) {
            int orow = rb + r + half * 8;
            float v = fmaxf(acc[ct][r] * sc + bi, 0.f) + x[(size_t)orow * COUT + c];
            out[(size_t)orow * COUT + c] = v;
        }
    }
}

// ---------------- host launcher ----------------
extern "C" void kernel_launch(void* const* d_in, const int* in_sizes, int n_in,
                              void* d_out, int out_size, void* d_ws, size_t ws_size,
                              hipStream_t stream) {
    const float* x  = (const float*)d_in[0];
    const float* W1 = (const float*)d_in[1];
    const float* g1 = (const float*)d_in[2];
    const float* b1 = (const float*)d_in[3];
    const float* Wk = (const float*)d_in[4];
    const float* W2 = (const float*)d_in[5];
    const float* g2 = (const float*)d_in[6];
    const float* b2 = (const float*)d_in[7];
    const int*  nbr = (const int*)d_in[8];
    float* out = (float*)d_out;

    char* ws = (char*)d_ws;
    float*          stats = (float*)(ws + OFF_STATS);
    unsigned short* wkT   = (unsigned short*)(ws + OFF_WKT);
    unsigned short* h1    = (unsigned short*)(ws + OFF_H1);
    float*          yA    = (float*)(ws + OFF_YA);     // y1 f32, later reused as h2 f32

    const int TILES = NVOX / 128;        // 3125 workgroups: 8 waves x 16 rows

    k_zero   <<<3, 256, 0, stream>>>(stats);
    k_cvt_wkT<<<(KOFF * 64 * 64 + 255) / 256, 256, 0, stream>>>(Wk, wkT);
    k_gemm1  <<<TILES, 256, 0, stream>>>(x, W1, yA, stats);
    k_fin1   <<<1, 64, 0, stream>>>(g1, b1, stats);
    k_bn1    <<<(NVOX * D2 / 4 + 255) / 256, 256, 0, stream>>>(yA, stats, h1);
    k_spconv <<<TILES, 256, 0, stream>>>(h1, wkT, nbr, yA);          // yA now holds h2
    k_gemm2_stats<<<TILES, 256, 0, stream>>>(yA, W2, stats);
    k_fin2   <<<1, 256, 0, stream>>>(g2, b2, stats);
    k_final  <<<TILES, 256, 0, stream>>>(yA, W2, x, stats, out);
}
